// Decoder_24799141167264
// MI455X (gfx1250) — matrix operand order repaired and checked by reading, not compiled, not run on hardware
//
#include <hip/hip_runtime.h>
#include <cmath>

#define B_   32
#define L_   2048
#define H_   512
#define V_   32000
#define TWOH 1024
#define G3H  1536
#define LCH  16          // chunks over L for the mean reduction
#define RPC  (L_/LCH)    // rows per chunk = 128
#define TROWS 16         // rows per LDS tile (16*1024*4 = 64KB)

// ---- CDNA5 async global->LDS path (guarded; falls back to plain loads) ----
#if defined(__HIP_DEVICE_COMPILE__) && \
    __has_builtin(__builtin_amdgcn_global_load_async_to_lds_b128) && \
    __has_builtin(__builtin_amdgcn_s_wait_asynccnt)
#define USE_ASYNC 1
typedef int v4i_ __attribute__((vector_size(4 * sizeof(int))));
typedef __attribute__((address_space(1))) v4i_ GV4;   // global <4 x i32>
typedef __attribute__((address_space(3))) v4i_ LV4;   // LDS    <4 x i32>
#else
#define USE_ASYNC 0
#endif

typedef _Float16 half16 __attribute__((ext_vector_type(16)));
typedef float    f32x8  __attribute__((ext_vector_type(8)));

// -------------------------------------------------------------------------
// attn_weights = softmax over constant betas = 1/L everywhere
// -------------------------------------------------------------------------
__global__ void fill_attn(float* __restrict__ out) {
    int i = blockIdx.x * 256 + threadIdx.x;       // grid covers B*L exactly
    out[i] = 1.0f / (float)L_;
}

// -------------------------------------------------------------------------
// Partial column sums of encoder_out over L (streams 256MB). Each block owns
// (b, chunk of 128 rows); tiles of 16 rows x 1024 f32 are staged through LDS
// with async global->LDS loads when available.
// -------------------------------------------------------------------------
__global__ void mean_partial(const float* __restrict__ enc,
                             float* __restrict__ partial) {
    __shared__ float tile[TROWS * TWOH];          // 64 KB
    const int b  = blockIdx.x / LCH;
    const int ch = blockIdx.x % LCH;
    const int t  = threadIdx.x;                   // 256 threads
    const float* base = enc + ((size_t)b * L_ + (size_t)ch * RPC) * TWOH;

    float4 acc = make_float4(0.f, 0.f, 0.f, 0.f);
    for (int tl = 0; tl < RPC / TROWS; ++tl) {    // 8 tiles
        const float* gsrc = base + (size_t)tl * TROWS * TWOH;
        for (int s = t; s < TROWS * TWOH / 4; s += 256) {
            int o = s * 4;
#if USE_ASYNC
            __builtin_amdgcn_global_load_async_to_lds_b128(
                (GV4*)(gsrc + o), (LV4*)(&tile[o]), 0, 0);
#else
            *(float4*)(&tile[o]) = *(const float4*)(gsrc + o);
#endif
        }
#if USE_ASYNC
        __builtin_amdgcn_s_wait_asynccnt(0);
#endif
        __syncthreads();
        const int c0 = t * 4;
#pragma unroll
        for (int r = 0; r < TROWS; ++r) {
            float4 v = *(const float4*)(&tile[r * TWOH + c0]);
            acc.x += v.x; acc.y += v.y; acc.z += v.z; acc.w += v.w;
        }
        __syncthreads();
    }
    *(float4*)(&partial[(size_t)blockIdx.x * TWOH + t * 4]) = acc;
}

// c[b][:] = relu(sum / L)
__global__ void mean_reduce(const float* __restrict__ partial,
                            float* __restrict__ c) {
    const int b = blockIdx.x, c0 = threadIdx.x * 4;
    float4 s = make_float4(0.f, 0.f, 0.f, 0.f);
    for (int ch = 0; ch < LCH; ++ch) {
        float4 v = *(const float4*)(&partial[((size_t)b * LCH + ch) * TWOH + c0]);
        s.x += v.x; s.y += v.y; s.z += v.z; s.w += v.w;
    }
    const float inv = 1.0f / (float)L_;
    float4 o;
    o.x = fmaxf(s.x * inv, 0.f); o.y = fmaxf(s.y * inv, 0.f);
    o.z = fmaxf(s.z * inv, 0.f); o.w = fmaxf(s.w * inv, 0.f);
    *(float4*)(&c[(size_t)b * TWOH + c0]) = o;
}

// -------------------------------------------------------------------------
// WMMA GEMM: Out[M x N] = A[M x K] @ W[N x K]^T   (f32 in/out, f16 WMMA)
// One wave per 16x16 output tile; K stepped by 32 per v_wmma_f32_16x16x32_f16.
// Register layouts per CDNA5 ISA 7.12.2.
// -------------------------------------------------------------------------
__global__ void wmma_gemm_nt(const float* __restrict__ A,
                             const float* __restrict__ W,
                             float* __restrict__ Out,
                             int M, int N, int K) {
    const int lane = threadIdx.x & 31;
    const int nt = blockIdx.x, mt = blockIdx.y;
    const int hs = lane >> 4;            // half-select (lanes 16-31)
    const int lm = lane & 15;
    const int am = mt * 16 + lm;         // A row this lane carries
    const int bn = nt * 16 + lm;         // W row (output column)
    const bool arow_ok = (am < M);
    const float* Arow = A + (size_t)am * K;
    const float* Wrow = W + (size_t)bn * K;

    f32x8 acc = {};
    for (int k0 = 0; k0 < K; k0 += 32) {
        // A 16x32 f16: elem j<8 -> K=k0+8*hs+j ; elem j>=8 -> K=k0+8*hs+16+(j-8)
        half16 a = {};
        if (arow_ok) {
            const int ka = k0 + hs * 8;
#pragma unroll
            for (int j = 0; j < 8; j += 2) {
                float2 v = *(const float2*)(Arow + ka + j);
                a[j] = (_Float16)v.x; a[j + 1] = (_Float16)v.y;
            }
#pragma unroll
            for (int j = 8; j < 16; j += 2) {
                float2 v = *(const float2*)(Arow + ka + 16 + (j - 8));
                a[j] = (_Float16)v.x; a[j + 1] = (_Float16)v.y;
            }
        }
        // B 32x16 f16: lane carries column bn; elem j<8 -> K=k0+8*hs+j ; elem j>=8 -> K=k0+8*hs+16+(j-8)
        half16 bvec;
        const int kb = k0 + hs * 8;
#pragma unroll
        for (int j = 0; j < 16; j += 4) {
            float4 v = *(const float4*)(Wrow + kb + j + (j >= 8 ? 8 : 0));
            bvec[j]     = (_Float16)v.x; bvec[j + 1] = (_Float16)v.y;
            bvec[j + 2] = (_Float16)v.z; bvec[j + 3] = (_Float16)v.w;
        }
        acc = __builtin_amdgcn_wmma_f32_16x16x32_f16(
            false, a, false, bvec, (short)0, acc, false, false);
    }
    // D: elem i -> row mt*16 + i + 8*hs, col nt*16 + lm
#pragma unroll
    for (int i = 0; i < 8; ++i) {
        int r = mt * 16 + i + hs * 8;
        if (r < M) Out[(size_t)r * N + nt * 16 + lm] = acc[i];
    }
}

// -------------------------------------------------------------------------
// GRU pointwise for both directions; also extracts c_out row 31 for logits.
// -------------------------------------------------------------------------
__global__ void gru_pointwise(const float* __restrict__ gi_f, const float* __restrict__ gh_f,
                              const float* __restrict__ gi_b, const float* __restrict__ gh_b,
                              const float* __restrict__ bi_f, const float* __restrict__ bh_f,
                              const float* __restrict__ bi_b, const float* __restrict__ bh_b,
                              const float* __restrict__ h_prev,
                              float* __restrict__ h_new, float* __restrict__ clast) {
    int idx = blockIdx.x * 256 + threadIdx.x;     // 2*32*512
    int dir = idx >> 14;
    int m   = (idx >> 9) & 31;
    int j   = idx & 511;
    const float* gi = dir ? gi_b : gi_f;
    const float* gh = dir ? gh_b : gh_f;
    const float* bi = dir ? bi_b : bi_f;
    const float* bh = dir ? bh_b : bh_f;
    float ir = gi[m * G3H + j]            + bi[j];
    float iz = gi[m * G3H + H_ + j]       + bi[H_ + j];
    float in_ = gi[m * G3H + 2 * H_ + j]  + bi[2 * H_ + j];
    float hr = gh[m * G3H + j]            + bh[j];
    float hz = gh[m * G3H + H_ + j]       + bh[H_ + j];
    float hn = gh[m * G3H + 2 * H_ + j]   + bh[2 * H_ + j];
    float r = 1.f / (1.f + __expf(-(ir + hr)));
    float z = 1.f / (1.f + __expf(-(iz + hz)));
    float n = tanhf(in_ + r * hn);
    float ho = (1.f - z) * n + z * h_prev[idx];
    h_new[idx] = ho;
    if (m == 31) clast[dir * H_ + j] = ho;        // c_out[-1] = [h_f[31], h_b[31]]
}

// -------------------------------------------------------------------------
// log_softmax over 32000 logits (+ bias), single 1024-thread block.
// -------------------------------------------------------------------------
__global__ void logsoftmax_k(const float* __restrict__ logits,
                             const float* __restrict__ bias,
                             float* __restrict__ out) {
    __shared__ float red[1024];
    const int t = threadIdx.x;
    float mx = -3.4e38f;
    for (int i = t; i < V_; i += 1024) mx = fmaxf(mx, logits[i] + bias[i]);
    red[t] = mx; __syncthreads();
    for (int off = 512; off > 0; off >>= 1) {
        if (t < off) red[t] = fmaxf(red[t], red[t + off]);
        __syncthreads();
    }
    float gmax = red[0]; __syncthreads();
    float sm = 0.f;
    for (int i = t; i < V_; i += 1024) sm += __expf(logits[i] + bias[i] - gmax);
    red[t] = sm; __syncthreads();
    for (int off = 512; off > 0; off >>= 1) {
        if (t < off) red[t] += red[t + off];
        __syncthreads();
    }
    float lse = gmax + __logf(red[0]);
    for (int i = t; i < V_; i += 1024) out[i] = logits[i] + bias[i] - lse;
}

// -------------------------------------------------------------------------
extern "C" void kernel_launch(void* const* d_in, const int* in_sizes, int n_in,
                              void* d_out, int out_size, void* d_ws, size_t ws_size,
                              hipStream_t stream) {
    (void)in_sizes; (void)n_in; (void)out_size; (void)ws_size;
    const float* h      = (const float*)d_in[1];   // (2,32,512)
    const float* enc    = (const float*)d_in[2];   // (32,2048,1024)
    const float* W_ih_f = (const float*)d_in[6];
    const float* W_hh_f = (const float*)d_in[7];
    const float* b_ih_f = (const float*)d_in[8];
    const float* b_hh_f = (const float*)d_in[9];
    const float* W_ih_b = (const float*)d_in[10];
    const float* W_hh_b = (const float*)d_in[11];
    const float* b_ih_b = (const float*)d_in[12];
    const float* b_hh_b = (const float*)d_in[13];
    const float* out_W  = (const float*)d_in[14];  // (32000,1024)
    const float* out_b  = (const float*)d_in[15];

    float* ws      = (float*)d_ws;
    float* partial = ws;                           // 32*16*1024
    float* c       = partial + B_ * LCH * TWOH;    // 32*1024
    float* gi_f    = c + B_ * TWOH;                // 32*1536 each
    float* gh_f    = gi_f + B_ * G3H;
    float* gi_b    = gh_f + B_ * G3H;
    float* gh_b    = gi_b + B_ * G3H;
    float* clast   = gh_b + B_ * G3H;              // 1024
    float* logits  = clast + TWOH;                 // 32000

    float* out_logp = (float*)d_out;               // 32000
    float* out_h    = out_logp + V_;               // 2*32*512
    float* out_attn = out_h + 2 * B_ * H_;         // 32*2048

    fill_attn   <<<dim3(B_ * L_ / 256), dim3(256), 0, stream>>>(out_attn);
    mean_partial<<<dim3(B_ * LCH),      dim3(256), 0, stream>>>(enc, partial);
    mean_reduce <<<dim3(B_),            dim3(256), 0, stream>>>(partial, c);

    wmma_gemm_nt<<<dim3(G3H / 16, 2), dim3(32), 0, stream>>>(c,           W_ih_f, gi_f, B_, G3H, TWOH);
    wmma_gemm_nt<<<dim3(G3H / 16, 2), dim3(32), 0, stream>>>(h,           W_hh_f, gh_f, B_, G3H, H_);
    wmma_gemm_nt<<<dim3(G3H / 16, 2), dim3(32), 0, stream>>>(c,           W_ih_b, gi_b, B_, G3H, TWOH);
    wmma_gemm_nt<<<dim3(G3H / 16, 2), dim3(32), 0, stream>>>(h + B_ * H_, W_hh_b, gh_b, B_, G3H, H_);

    gru_pointwise<<<dim3(2 * B_ * H_ / 256), dim3(256), 0, stream>>>(
        gi_f, gh_f, gi_b, gh_b, b_ih_f, b_hh_f, b_ih_b, b_hh_b, h, out_h, clast);

    // logits matvec padded to M=16 (row 0 live), bandwidth-bound on out_W
    wmma_gemm_nt<<<dim3(V_ / 16, 1), dim3(32), 0, stream>>>(clast, out_W, logits, 1, V_, TWOH);

    logsoftmax_k<<<dim3(1), dim3(1024), 0, stream>>>(logits, out_b, out_logp);
}
